// DSConv_pro_72507637891270
// MI455X (gfx1250) — matrix operand order repaired and checked by reading, not compiled, not run on hardware
//
#include <hip/hip_runtime.h>
#include <hip/hip_bf16.h>

typedef __attribute__((ext_vector_type(16))) _Float16 v16h;
typedef __attribute__((ext_vector_type(2)))  _Float16 v2h;
typedef __attribute__((ext_vector_type(8)))  float    v8f;

#define NB 4
#define NC 64
#define NO 64
#define NH 128
#define NW 128
#define NK 9          // snake kernel taps
#define OC2 18        // offset-conv output channels (2*K)
#define LDS_STRIDE 72 // halves per w-row in LDS staging (16B aligned, bank-spread)

__device__ __forceinline__ int iclamp(int v, int lo, int hi) {
    return v < lo ? lo : (v > hi ? hi : v);
}

// ---------------------------------------------------------------------------
// Prep: convert w_off / w_dsc (f32) into per-lane WMMA A-fragments (f16).
// Fragment index: ((((mt*9 + tap)*2 + kc)*32 + lane)*16 + j)
//   M  = mt*16 + (lane & 15)
//   kk = (lane<16 ? 0 : 16) + j   (within a 32-wide K chunk)
//   c  = kc*32 + kk
// ---------------------------------------------------------------------------
__global__ void prep_frags_kernel(const float* __restrict__ w_off,
                                  const float* __restrict__ w_dsc,
                                  _Float16* __restrict__ afc,
                                  _Float16* __restrict__ afd) {
    int idx = blockIdx.x * 256 + threadIdx.x;
    // conv A-frags: 2 mtiles * 9 taps * 2 kc * 32 lanes * 16 halves = 18432
    if (idx < 18432) {
        int j    = idx & 15;
        int lane = (idx >> 4) & 31;
        int kc   = (idx >> 9) & 1;
        int rest = idx >> 10;
        int tap  = rest % 9;
        int mt   = rest / 9;
        int M    = mt * 16 + (lane & 15);
        int kk   = ((lane < 16) ? 0 : 16) + j;
        int c    = kc * 32 + kk;
        int dy   = tap / 3, dx = tap % 3;
        float v  = 0.f;
        if (M < OC2) v = w_off[((M * NC + c) * 3 + dy) * 3 + dx];
        afc[idx] = (_Float16)v;
    }
    int idx2 = idx - 18432;
    // dsc A-frags: 4 mtiles * 9 * 2 * 32 * 16 = 36864
    if (idx2 >= 0 && idx2 < 36864) {
        int j    = idx2 & 15;
        int lane = (idx2 >> 4) & 31;
        int kc   = (idx2 >> 9) & 1;
        int rest = idx2 >> 10;
        int tap  = rest % 9;
        int mt   = rest / 9;
        int M    = mt * 16 + (lane & 15);
        int kk   = ((lane < 16) ? 0 : 16) + j;
        int c    = kc * 32 + kk;
        afd[idx2] = (_Float16)w_dsc[(M * NC + c) * NK + tap];
    }
}

// ---------------------------------------------------------------------------
// Offset conv 3x3 (C=64 -> 18) as im2col + WMMA GEMM per (b,h).
// M = 32 (2 tiles, rows 18..31 padded zero), Kdim = 9 taps * 64 c, N = 128 w.
// ---------------------------------------------------------------------------
__global__ __launch_bounds__(256) void conv_off_kernel(
        const float* __restrict__ x, const float* __restrict__ b_off,
        const _Float16* __restrict__ afc, float* __restrict__ off_raw) {
    __shared__ __align__(32) _Float16 dbuf[2][NW * LDS_STRIDE];
    const int h = blockIdx.x, b = blockIdx.y;
    const int tid = threadIdx.x, lane = tid & 31, wv = tid >> 5;
    const float* xb = x + b * (NC * NH * NW);   // per-batch base (saddr)

    // Branchless staging, 32-bit element offsets -> saddr+voffset loads.
    // Each thread handles channels (c2, c2+1) at one w -> 1 packed b32 DS store.
    auto stage = [&](int tap, int buf) {
        int dy = tap / 3 - 1, dx = tap % 3 - 1;
        int hy = h + dy;
        const bool hok = (hy >= 0) && (hy < NH);
        const int hyc  = iclamp(hy, 0, NH - 1);
        #pragma unroll 4
        for (int it = 0; it < 16; ++it) {
            int i  = it * 256 + tid;        // 0..4095 over (c-pair, w)
            int w  = i & 127;
            int c2 = (i >> 7) << 1;         // 0,2,...,62
            int wx = w + dx;
            bool ok = hok && (wx >= 0) && (wx < NW);
            int wxc  = iclamp(wx, 0, NW - 1);
            int idx0 = (c2 * NH + hyc) * NW + wxc;   // 32-bit element offset
            float v0 = xb[idx0];
            float v1 = xb[idx0 + NH * NW];
            v0 = ok ? v0 : 0.f;
            v1 = ok ? v1 : 0.f;
            v2h pk = { (_Float16)v0, (_Float16)v1 };
            *(v2h*)&dbuf[buf][w * LDS_STRIDE + c2] = pk;
        }
    };

    stage(0, 0);
    __syncthreads();

    const int mt  = wv >> 2;        // 0..1 : oc tile
    const int nt0 = (wv & 3) * 2;   // N tiles nt0, nt0+1 (16 w each)
    const int wlo = lane & 15;
    const int cbh = (lane < 16) ? 0 : 16;
    v8f acc0 = {}, acc1 = {};

    for (int tap = 0; tap < NK; ++tap) {
        int buf = tap & 1;
        #pragma unroll
        for (int kc = 0; kc < 2; ++kc) {
            v16h a = *(const v16h*)(afc + ((((mt * 9 + tap) * 2 + kc) * 32 + lane) << 4));
            int cb = kc * 32 + cbh;
            v16h b0 = *(const v16h*)(&dbuf[buf][(nt0 * 16 + wlo) * LDS_STRIDE + cb]);
            v16h b1 = *(const v16h*)(&dbuf[buf][((nt0 + 1) * 16 + wlo) * LDS_STRIDE + cb]);
            acc0 = __builtin_amdgcn_wmma_f32_16x16x32_f16(false, a, false, b0,
                                                          (short)0, acc0, false, false);
            acc1 = __builtin_amdgcn_wmma_f32_16x16x32_f16(false, a, false, b1,
                                                          (short)0, acc1, false, false);
        }
        if (tap < NK - 1) stage(tap + 1, (tap + 1) & 1);
        __syncthreads();
    }

    const int mro = (lane < 16) ? 0 : 8;
    #pragma unroll
    for (int r = 0; r < 8; ++r) {
        int oc = mt * 16 + r + mro;
        if (oc < OC2) {
            float bias = b_off[oc];
            float* dst = off_raw + ((b * OC2 + oc) * NH + h) * NW;
            dst[nt0 * 16 + wlo]       = acc0[r] + bias;
            dst[(nt0 + 1) * 16 + wlo] = acc1[r] + bias;
        }
    }
}

// ---------------------------------------------------------------------------
// GroupNorm statistics: one workgroup per (b, group).
// ---------------------------------------------------------------------------
__global__ __launch_bounds__(256) void gn_stats_kernel(
        const float* __restrict__ src, float* __restrict__ stats,
        int ch_per_grp, int n_grp_per_b, int nch) {
    int g = blockIdx.x % n_grp_per_b;
    int b = blockIdx.x / n_grp_per_b;
    const float* p = src + (b * nch + g * ch_per_grp) * (NH * NW);
    int n = ch_per_grp * NH * NW;
    float s = 0.f, s2 = 0.f;
    #pragma unroll 4
    for (int i = threadIdx.x; i < n; i += 256) {
        float v = p[i];
        s += v; s2 += v * v;
    }
    __shared__ float sh[256], sh2[256];
    sh[threadIdx.x] = s; sh2[threadIdx.x] = s2;
    __syncthreads();
    for (int st = 128; st > 0; st >>= 1) {
        if (threadIdx.x < st) {
            sh[threadIdx.x]  += sh[threadIdx.x + st];
            sh2[threadIdx.x] += sh2[threadIdx.x + st];
        }
        __syncthreads();
    }
    if (threadIdx.x == 0) {
        float inv_n = 1.f / (float)n;
        float mu = sh[0] * inv_n;
        float var = sh2[0] * inv_n - mu * mu;
        stats[blockIdx.x * 2]     = mu;
        stats[blockIdx.x * 2 + 1] = rsqrtf(var + 1e-5f);
    }
}

// ---------------------------------------------------------------------------
// GN + tanh + directional cumsum -> ymap[b,k,h,w] (clipped sample row).
// ---------------------------------------------------------------------------
__global__ __launch_bounds__(256) void ymap_kernel(
        const float* __restrict__ off_raw, const float* __restrict__ stats_off,
        const float* __restrict__ gg, const float* __restrict__ gb,
        float* __restrict__ ymap) {
    int idx = blockIdx.x * 256 + threadIdx.x; // over b*h*w
    if (idx >= NB * NH * NW) return;
    int w = idx & 127;
    int h = (idx >> 7) & 127;
    int b = idx >> 14;
    float v[NK];
    #pragma unroll
    for (int ch = 0; ch < NK; ++ch) {
        int g = ch >> 1; // groups of 2 channels (18 ch / 9 groups)
        float mu   = stats_off[(b * NK + g) * 2];
        float rinv = stats_off[(b * NK + g) * 2 + 1];
        float t = off_raw[((b * OC2 + ch) * NH + h) * NW + w];
        v[ch] = tanhf((t - mu) * rinv * gg[ch] + gb[ch]);
    }
    float yn[NK];
    yn[4] = 0.f;
    float s = 0.f;
    #pragma unroll
    for (int k = 3; k >= 0; --k) { s += v[k]; yn[k] = s; }   // down
    s = 0.f;
    #pragma unroll
    for (int k = 5; k < NK; ++k) { s += v[k]; yn[k] = s; }   // up
    #pragma unroll
    for (int k = 0; k < NK; ++k) {
        float ym = (float)h + yn[k];                         // EXTEND = 1.0
        ym = fminf(fmaxf(ym, 0.f), (float)(NH - 1));
        ymap[((b * NK + k) * NH + h) * NW + w] = ym;
    }
}

// ---------------------------------------------------------------------------
// Fused deformable-sample + einsum('bchkw,ock->bohw') via WMMA per (b,h).
// x_new is integer -> sampling is 1D linear interp in y only; interpolated
// f16 tap slices are staged in LDS and never materialized in HBM.
// M = 64 (o), Kdim = 9 taps * 64 c = 576, N = 128 (w).
// ---------------------------------------------------------------------------
__global__ __launch_bounds__(256) void deform_gemm_kernel(
        const float* __restrict__ x, const float* __restrict__ ymap,
        const float* __restrict__ b_dsc, const _Float16* __restrict__ afd,
        float* __restrict__ out_raw) {
    __shared__ __align__(32) _Float16 dbuf[2][NW * LDS_STRIDE];
    __shared__ int2 s_yw[NK * NW];   // .x = y0 row offset (elements), .y = bitcast(wy)
    const int h = blockIdx.x, b = blockIdx.y;
    const int tid = threadIdx.x, lane = tid & 31, wv = tid >> 5;
    const float* xb = x + b * (NC * NH * NW);

    for (int i = tid; i < NK * NW; i += 256) {
        int k = i >> 7, w = i & 127;
        float ym = ymap[((b * NK + k) * NH + h) * NW + w];
        int y0 = iclamp((int)floorf(ym), 0, NH - 1);
        float wy = ym - (float)y0;
        // pack: row0 element offset, y1 delta folded into sign-free select later
        s_yw[i] = make_int2(y0 * NW, __float_as_int(wy));
    }
    __syncthreads();

    // Branchless interp staging; channels (c2, c2+1) per thread, packed store.
    auto stage = [&](int tap, int buf) {
        int dx = tap - 4;
        #pragma unroll 4
        for (int it = 0; it < 16; ++it) {
            int i  = it * 256 + tid;
            int w  = i & 127;
            int c2 = (i >> 7) << 1;
            int xi = iclamp(w + dx, 0, NW - 1);
            int2 yw = s_yw[tap * NW + w];
            int row0 = yw.x;                          // y0*NW
            float wy = __int_as_float(yw.y);
            int dy1  = (row0 < (NH - 1) * NW) ? NW : 0; // y1 = min(y0+1, NH-1)
            int idx0 = c2 * (NH * NW) + row0 + xi;    // 32-bit element offset
            float a0 = xb[idx0];
            float b0 = xb[idx0 + dy1];
            float a1 = xb[idx0 + NH * NW];
            float b1 = xb[idx0 + NH * NW + dy1];
            float r0 = a0 + wy * (b0 - a0);
            float r1 = a1 + wy * (b1 - a1);
            v2h pk = { (_Float16)r0, (_Float16)r1 };
            *(v2h*)&dbuf[buf][w * LDS_STRIDE + c2] = pk;
        }
    };

    stage(0, 0);
    __syncthreads();

    const int mt  = wv >> 1;       // 0..3 : o tile
    const int nh  = wv & 1;        // 0..1 : 64-wide w half
    const int wlo = lane & 15;
    const int cbh = (lane < 16) ? 0 : 16;
    v8f acc[4] = {{}, {}, {}, {}};

    for (int tap = 0; tap < NK; ++tap) {
        int buf = tap & 1;
        #pragma unroll
        for (int kc = 0; kc < 2; ++kc) {
            v16h a = *(const v16h*)(afd + ((((mt * 9 + tap) * 2 + kc) * 32 + lane) << 4));
            int cb = kc * 32 + cbh;
            #pragma unroll
            for (int nt = 0; nt < 4; ++nt) {
                int w = (nh * 4 + nt) * 16 + wlo;
                v16h bm = *(const v16h*)(&dbuf[buf][w * LDS_STRIDE + cb]);
                acc[nt] = __builtin_amdgcn_wmma_f32_16x16x32_f16(false, a, false, bm,
                                                                 (short)0, acc[nt],
                                                                 false, false);
            }
        }
        if (tap < NK - 1) stage(tap + 1, (tap + 1) & 1);
        __syncthreads();
    }

    const int mro = (lane < 16) ? 0 : 8;
    #pragma unroll
    for (int nt = 0; nt < 4; ++nt) {
        int w = (nh * 4 + nt) * 16 + wlo;
        #pragma unroll
        for (int r = 0; r < 8; ++r) {
            int o = mt * 16 + r + mro;
            out_raw[((b * NO + o) * NH + h) * NW + w] = acc[nt][r] + b_dsc[o];
        }
    }
}

// ---------------------------------------------------------------------------
// Final GroupNorm(16 groups of 4 ch) + ReLU.
// ---------------------------------------------------------------------------
__global__ __launch_bounds__(256) void final_gn_relu_kernel(
        const float* __restrict__ out_raw, const float* __restrict__ stats,
        const float* __restrict__ gg, const float* __restrict__ gb,
        float* __restrict__ out) {
    int idx = blockIdx.x * 256 + threadIdx.x;
    if (idx >= NB * NO * NH * NW) return;
    int o = (idx >> 14) & 63;
    int b = idx >> 20;
    int g = o >> 2;
    float mu   = stats[(b * 16 + g) * 2];
    float rinv = stats[(b * 16 + g) * 2 + 1];
    float v = (out_raw[idx] - mu) * rinv * gg[o] + gb[o];
    out[idx] = fmaxf(v, 0.f);
}

// ---------------------------------------------------------------------------
extern "C" void kernel_launch(void* const* d_in, const int* in_sizes, int n_in,
                              void* d_out, int out_size, void* d_ws, size_t ws_size,
                              hipStream_t stream) {
    (void)in_sizes; (void)n_in; (void)out_size; (void)ws_size;
    const float* x        = (const float*)d_in[0];
    const float* w_off    = (const float*)d_in[1];
    const float* b_off    = (const float*)d_in[2];
    const float* gn_off_g = (const float*)d_in[3];
    const float* gn_off_b = (const float*)d_in[4];
    const float* w_dsc    = (const float*)d_in[5];
    const float* b_dsc    = (const float*)d_in[6];
    const float* gn_g     = (const float*)d_in[7];
    const float* gn_b     = (const float*)d_in[8];
    float* out = (float*)d_out;

    char* ws = (char*)d_ws;
    _Float16* afc = (_Float16*)ws;                 // 36864 B
    _Float16* afd = (_Float16*)(ws + 36864);       // 73728 B
    float* fbase    = (float*)(ws + 36864 + 73728);
    float* off_raw  = fbase;                       // 4*18*128*128   = 1179648 f
    float* ymap     = off_raw + 1179648;           // 4*9*128*128    =  589824 f
    float* out_raw  = ymap + 589824;               // 4*64*128*128   = 4194304 f
    float* stats_off = out_raw + 4194304;          // 36*2
    float* stats_out = stats_off + 72;             // 64*2

    prep_frags_kernel<<<(18432 + 36864 + 255) / 256, 256, 0, stream>>>(
        w_off, w_dsc, afc, afd);

    dim3 gbh(NH, NB);
    conv_off_kernel<<<gbh, 256, 0, stream>>>(x, b_off, afc, off_raw);

    gn_stats_kernel<<<NB * NK, 256, 0, stream>>>(off_raw, stats_off, 2, NK, OC2);

    ymap_kernel<<<(NB * NH * NW + 255) / 256, 256, 0, stream>>>(
        off_raw, stats_off, gn_off_g, gn_off_b, ymap);

    deform_gemm_kernel<<<gbh, 256, 0, stream>>>(x, ymap, b_dsc, afd, out_raw);

    gn_stats_kernel<<<NB * 16, 256, 0, stream>>>(out_raw, stats_out, 4, 16, NO);

    final_gn_relu_kernel<<<(NB * NO * NH * NW + 255) / 256, 256, 0, stream>>>(
        out_raw, stats_out, gn_g, gn_b, out);
}
